// VQVAE_70171175682488
// MI455X (gfx1250) — matrix operand order repaired and checked by reading, not compiled, not run on hardware
//
#include <hip/hip_runtime.h>
#include <math.h>

// ---------------------------------------------------------------------------
// VQ-VAE forward for MI455X (gfx1250, wave32).
// All matmuls (transformer QKV/proj/FFN, attention QK^T and PV, enc/dec linear,
// VQ distance matrix) run on v_wmma_f32_16x16x32_f16 (f32->f16 convert on load,
// f32 accumulate). Main GEMM: 64x64 register blocking per wave (4x4 tiles),
// __launch_bounds__(256,2), A panel loaded as a b128 clause and held as f16
// fragments, B panel software-pipelined over the 4 n-subtiles (next panel's 4
// b128 loads issued before the current panel's convert+WMMA) so load latency
// overlaps matrix work without exceeding the 256-VGPR direct-address window.
// Convs / LN / softmax / VQ-argmin are memory-bound VALU work at 23.3 TB/s.
// ---------------------------------------------------------------------------

typedef _Float16 v16h __attribute__((ext_vector_type(16)));
typedef float    v8f  __attribute__((ext_vector_type(8)));

#define DEVI static __device__ __forceinline__

// ---- wave-level 16x16 output tile GEMM, K multiple of 32 -------------------
// BT=true : B given as Bt (N,K) row-major (torch weight layout / K-matrix)
// BT=false: B given as (K,N) row-major with leading stride ldb
template <bool BT>
DEVI v8f wave_tile(const float* __restrict__ A, long lda,
                   const float* __restrict__ B, long ldb,
                   int K, int lane)
{
  v8f acc = {0.f, 0.f, 0.f, 0.f, 0.f, 0.f, 0.f, 0.f};
  const int half = lane >> 4;   // K-half selector (A/B), row-half selector (C)
  const int sub  = lane & 15;   // A: row m ; B: col n
  const float* Arow = A + (long)sub * lda;
  const float* Brow = BT ? (B + (long)sub * ldb) : B;
  for (int kb = 0; kb < K; kb += 32) {
    v16h a, b;
    {   // A 16x32: 16 contiguous floats at kb+half*16
      const float* ap = Arow + kb + (half << 4);
      float4 a0 = *(const float4*)(ap);
      float4 a1 = *(const float4*)(ap + 4);
      float4 a2 = *(const float4*)(ap + 8);
      float4 a3 = *(const float4*)(ap + 12);
      a[0]=(_Float16)a0.x; a[1]=(_Float16)a0.y; a[2]=(_Float16)a0.z; a[3]=(_Float16)a0.w;
      a[4]=(_Float16)a1.x; a[5]=(_Float16)a1.y; a[6]=(_Float16)a1.z; a[7]=(_Float16)a1.w;
      a[8]=(_Float16)a2.x; a[9]=(_Float16)a2.y; a[10]=(_Float16)a2.z; a[11]=(_Float16)a2.w;
      a[12]=(_Float16)a3.x; a[13]=(_Float16)a3.y; a[14]=(_Float16)a3.z; a[15]=(_Float16)a3.w;
    }
    if (BT) {  // B 32x16: 16 contiguous floats at kb + half*16
      const float* bp = Brow + kb + (half << 4);
#pragma unroll
      for (int q = 0; q < 4; ++q) {
        float4 f = *(const float4*)(bp + 4 * q);
        b[4*q]   = (_Float16)f.x; b[4*q+1] = (_Float16)f.y;
        b[4*q+2] = (_Float16)f.z; b[4*q+3] = (_Float16)f.w;
      }
    } else {
#pragma unroll
      for (int v = 0; v < 8; ++v) {
        int kk = kb + (half << 4) + (v << 1);
        b[2 * v]     = (_Float16)Brow[(long)kk * ldb + sub];
        b[2 * v + 1] = (_Float16)Brow[(long)(kk + 1) * ldb + sub];
      }
    }
    acc = __builtin_amdgcn_wmma_f32_16x16x32_f16(false, a, false, b,
                                                 (short)0, acc, false, false);
  }
  return acc;
}

// ---- register-blocked GEMM: C(M,N) = A(M,K) @ Bt(N,K)^T + bias -------------
// 64x64 output per wave: 4x4 grid of 16x16 WMMA tiles. M,N multiples of 64,
// K multiple of 32. Optional exact-erf GELU epilogue.
__global__ void __launch_bounds__(256, 2)
gemm_bias_act(const float* __restrict__ A,
              const float* __restrict__ Bt,
              const float* __restrict__ bias,
              float* __restrict__ C,
              int M, int N, int K, int act)
{
  int wid  = (blockIdx.x * blockDim.x + threadIdx.x) >> 5;
  int lane = threadIdx.x & 31;
  int ntiles = N >> 6;
  if (wid >= (M >> 6) * ntiles) return;          // wave-uniform exit
  int mt = wid / ntiles, nt = wid - mt * ntiles;
  const int half = lane >> 4, sub = lane & 15;

  v8f acc[4][4];
#pragma unroll
  for (int i = 0; i < 4; ++i)
#pragma unroll
    for (int j = 0; j < 4; ++j)
      acc[i][j] = (v8f){0.f, 0.f, 0.f, 0.f, 0.f, 0.f, 0.f, 0.f};

  const float* Arow = A  + ((long)mt * 64 + sub) * K;
  const float* Brow = Bt + ((long)nt * 64 + sub) * K;

  for (int kb = 0; kb < K; kb += 32) {
    if (kb + 288 <= K) {                         // prefetch 8 K-steps ahead
#pragma unroll
      for (int i = 0; i < 4; ++i) {
        __builtin_prefetch(Arow + (long)i * 16 * K + kb + 256, 0, 0);
        __builtin_prefetch(Brow + (long)i * 16 * K + kb + 256, 0, 0);
      }
    }
    // ---- A panel: 16 independent b128 loads (one clause), convert to f16 ---
    float4 ar[4][4];
#pragma unroll
    for (int i = 0; i < 4; ++i) {
      const float* ap = Arow + (long)i * 16 * K + kb + (half << 4);
      ar[i][0] = *(const float4*)(ap);
      ar[i][1] = *(const float4*)(ap + 4);
      ar[i][2] = *(const float4*)(ap + 8);
      ar[i][3] = *(const float4*)(ap + 12);
    }
    v16h af[4];
#pragma unroll
    for (int i = 0; i < 4; ++i) {
#pragma unroll
      for (int q = 0; q < 4; ++q) {
        af[i][4*q]   = (_Float16)ar[i][q].x;
        af[i][4*q+1] = (_Float16)ar[i][q].y;
        af[i][4*q+2] = (_Float16)ar[i][q].z;
        af[i][4*q+3] = (_Float16)ar[i][q].w;
      }
    }
    // ---- B panel: 2-deep software pipeline over the 4 n-subtiles -----------
    // Issue the next fragment's 4 b128 loads before converting/consuming the
    // current one, so load latency overlaps the 4 WMMAs + converts.
    float4 cur[4], nxt[4];
    {
      const float* bp0 = Brow + kb + (half << 4);
#pragma unroll
      for (int q = 0; q < 4; ++q) cur[q] = *(const float4*)(bp0 + 4 * q);
    }
#pragma unroll
    for (int j = 0; j < 4; ++j) {
      if (j < 3) {
        const float* bpn = Brow + (long)(j + 1) * 16 * K + kb + (half << 4);
#pragma unroll
        for (int q = 0; q < 4; ++q) nxt[q] = *(const float4*)(bpn + 4 * q);
      }
      v16h bf;
#pragma unroll
      for (int q = 0; q < 4; ++q) {
        bf[4*q]   = (_Float16)cur[q].x;
        bf[4*q+1] = (_Float16)cur[q].y;
        bf[4*q+2] = (_Float16)cur[q].z;
        bf[4*q+3] = (_Float16)cur[q].w;
      }
#pragma unroll
      for (int i = 0; i < 4; ++i)
        acc[i][j] = __builtin_amdgcn_wmma_f32_16x16x32_f16(false, af[i], false, bf,
                                                           (short)0, acc[i][j],
                                                           false, false);
      if (j < 3) {
#pragma unroll
        for (int q = 0; q < 4; ++q) cur[q] = nxt[q];
      }
    }
  }

#pragma unroll
  for (int j = 0; j < 4; ++j) {
    int col = nt * 64 + j * 16 + sub;
    float bv = bias ? bias[col] : 0.f;
#pragma unroll
    for (int i = 0; i < 4; ++i) {
#pragma unroll
      for (int v = 0; v < 8; ++v) {
        int row = mt * 64 + i * 16 + (half << 3) + v;
        float xv = acc[i][j][v] + bv;
        if (act) xv = 0.5f * xv * (1.f + erff(xv * 0.70710678118654752f));
        C[(long)row * N + col] = xv;
      }
    }
  }
}

// ---- attention: S = Q @ K^T per (b,h); QKV laid out (B*Tb, 2304) -----------
__global__ void attn_scores(const float* __restrict__ QKV,
                            float* __restrict__ S, int Tb)
{
  int b = blockIdx.z / 12, h = blockIdx.z % 12;
  int wid  = (blockIdx.x * blockDim.x + threadIdx.x) >> 5;
  int lane = threadIdx.x & 31;
  int tiles = Tb >> 4;
  if (wid >= tiles * tiles) return;
  int mt = wid / tiles, nt = wid - mt * tiles;
  const float* Q  = QKV + ((long)b * Tb + mt * 16) * 2304 + h * 64;
  const float* Kp = QKV + ((long)b * Tb + nt * 16) * 2304 + 768 + h * 64;
  v8f acc = wave_tile<true>(Q, 2304, Kp, 2304, 64, lane);
  float* Sr = S + (long)blockIdx.z * Tb * Tb;
  int half = lane >> 4, n = lane & 15;
#pragma unroll
  for (int v = 0; v < 8; ++v)
    Sr[(long)(mt * 16 + (half << 3) + v) * Tb + nt * 16 + n] = acc[v];
}

// ---- attention: O = P @ V per (b,h) ----------------------------------------
__global__ void attn_av(const float* __restrict__ S,
                        const float* __restrict__ QKV,
                        float* __restrict__ O, int Tb)
{
  int b = blockIdx.z / 12, h = blockIdx.z % 12;
  int wid  = (blockIdx.x * blockDim.x + threadIdx.x) >> 5;
  int lane = threadIdx.x & 31;
  int tiles = Tb >> 4;
  if (wid >= tiles * 4) return;                  // N = 64 -> 4 n-tiles
  int mt = wid >> 2, nt = wid & 3;
  const float* P = S + (long)blockIdx.z * Tb * Tb + (long)mt * 16 * Tb;
  const float* V = QKV + (long)b * Tb * 2304 + 1536 + h * 64 + nt * 16;
  v8f acc = wave_tile<false>(P, Tb, V, 2304, Tb, lane);
  int half = lane >> 4, n = lane & 15;
#pragma unroll
  for (int v = 0; v < 8; ++v)
    O[((long)b * Tb + mt * 16 + (half << 3) + v) * 768 + h * 64 + nt * 16 + n] = acc[v];
}

// ---- softmax over rows of length Tb, scale 1/sqrt(64), in place ------------
__global__ void softmax_rows(float* __restrict__ S, int Tb, int nrows)
{
  int row  = (blockIdx.x * blockDim.x + threadIdx.x) >> 5;
  int lane = threadIdx.x & 31;
  if (row >= nrows) return;
  float* p = S + (long)row * Tb;
  float m = -3.4e38f;
  for (int j = lane; j < Tb; j += 32) m = fmaxf(m, p[j]);
  for (int o = 16; o; o >>= 1) m = fmaxf(m, __shfl_xor(m, o, 32));
  float sum = 0.f;
  for (int j = lane; j < Tb; j += 32) {
    float e = expf((p[j] - m) * 0.125f);
    p[j] = e; sum += e;
  }
  for (int o = 16; o; o >>= 1) sum += __shfl_xor(sum, o, 32);
  float inv = 1.f / sum;
  for (int j = lane; j < Tb; j += 32) p[j] *= inv;
}

// ---- Out = LayerNorm(A + B) over D=768, one wave per row -------------------
__global__ void ln_add_768(const float* __restrict__ A, const float* __restrict__ Bs,
                           const float* __restrict__ g, const float* __restrict__ bt,
                           float* __restrict__ Out, int M)
{
  int row  = (blockIdx.x * blockDim.x + threadIdx.x) >> 5;
  int lane = threadIdx.x & 31;
  if (row >= M) return;
  const float* a = A  + (long)row * 768;
  const float* b = Bs + (long)row * 768;
  float vals[24], s = 0.f, s2 = 0.f;
#pragma unroll
  for (int i = 0; i < 24; ++i) {
    float v = a[lane + 32 * i] + b[lane + 32 * i];
    vals[i] = v; s += v; s2 += v * v;
  }
  for (int o = 16; o; o >>= 1) { s += __shfl_xor(s, o, 32); s2 += __shfl_xor(s2, o, 32); }
  float mean = s * (1.f / 768.f);
  float var  = s2 * (1.f / 768.f) - mean * mean;
  float inv  = rsqrtf(var + 1e-5f);
  float* o = Out + (long)row * 768;
#pragma unroll
  for (int i = 0; i < 24; ++i) {
    int c = lane + 32 * i;
    o[c] = (vals[i] - mean) * inv * g[c] + bt[c];
  }
}

// ---- direct conv1d k=5 pad=2, strided I/O, optional lrelu+eval-BN epilogue -
__global__ void conv1d_k5(const float* __restrict__ in, long sIb, long sIc, long sIt,
                          int Cin, int Tin,
                          const float* __restrict__ w, const float* __restrict__ wb,
                          const float* __restrict__ bng, const float* __restrict__ bnb,
                          float* __restrict__ out, long sOb, long sOc, long sOt,
                          int Cout, int Tout, int stride, int zeropad, int nB)
{
  int idx = blockIdx.x * blockDim.x + threadIdx.x;
  int total = nB * Cout * Tout;
  if (idx >= total) return;
  int t  = idx % Tout;
  int co = (idx / Tout) % Cout;
  int b  = idx / (Tout * Cout);
  const float* wr = w + (long)co * Cin * 5;
  float acc = wb[co];
  int t0 = t * stride - 2;
  for (int ci = 0; ci < Cin; ++ci) {
    const float* ip = in + (long)b * sIb + (long)ci * sIc;
#pragma unroll
    for (int k = 0; k < 5; ++k) {
      int ti = t0 + k;
      float xv;
      if (zeropad) xv = (ti < 0 || ti >= Tin) ? 0.f : ip[(long)ti * sIt];
      else { ti = ti < 0 ? 0 : (ti > Tin - 1 ? Tin - 1 : ti); xv = ip[(long)ti * sIt]; }
      acc = fmaf(wr[ci * 5 + k], xv, acc);
    }
  }
  if (bng) {
    acc = acc >= 0.f ? acc : 0.2f * acc;
    acc = acc * (bng[co] * rsqrtf(1.f + 1e-5f)) + bnb[co];
  }
  out[(long)b * sOb + (long)co * sOc + (long)t * sOt] = acc;
}

// ---- ConvTranspose1d k=5 stride=2 pad=2 outpad=1, w (Cin,Cout,5), +lrelu+BN -
__global__ void convT_k5_s2(const float* __restrict__ in, int Cin, int Tin,
                            const float* __restrict__ w, const float* __restrict__ wb,
                            const float* __restrict__ bng, const float* __restrict__ bnb,
                            float* __restrict__ out, int Cout, int Tout, int nB)
{
  int idx = blockIdx.x * blockDim.x + threadIdx.x;
  int total = nB * Cout * Tout;
  if (idx >= total) return;
  int t  = idx % Tout;
  int co = (idx / Tout) % Cout;
  int b  = idx / (Tout * Cout);
  float acc = wb[co];
  int imin = (t - 2) / 2 - 1; if (imin < 0) imin = 0;
  int imax = (t + 2) / 2;     if (imax > Tin - 1) imax = Tin - 1;
  for (int ci = 0; ci < Cin; ++ci) {
    const float* ip = in + ((long)b * Cin + ci) * Tin;
    const float* wr = w + ((long)ci * Cout + co) * 5;
    for (int i = imin; i <= imax; ++i) {
      int k = t + 2 - 2 * i;                 // k = t + pad - i*stride
      if (k >= 0 && k < 5) acc = fmaf(wr[k], ip[i], acc);
    }
  }
  acc = acc >= 0.f ? acc : 0.2f * acc;
  acc = acc * (bng[co] * rsqrtf(1.f + 1e-5f)) + bnb[co];
  out[((long)b * Cout + co) * Tout + t] = acc;
}

__global__ void maxpool2(const float* __restrict__ in, float* __restrict__ out,
                         int C, int Tout, int nB)
{
  int idx = blockIdx.x * blockDim.x + threadIdx.x;
  if (idx >= nB * C * Tout) return;
  int t = idx % Tout; int c = (idx / Tout) % C; int b = idx / (Tout * C);
  const float* p = in + ((long)b * C + c) * (2 * Tout);
  out[((long)b * C + c) * Tout + t] = fmaxf(p[2 * t], p[2 * t + 1]);
}

__global__ void repeat2(const float* __restrict__ in, float* __restrict__ out,
                        int C, int Tin, int nB)
{
  int idx = blockIdx.x * blockDim.x + threadIdx.x;
  if (idx >= nB * C * Tin) return;
  int t = idx % Tin; int c = (idx / Tin) % C; int b = idx / (Tin * C);
  float v = in[((long)b * C + c) * Tin + t];
  float* o = out + ((long)b * C + c) * (2 * Tin);
  o[2 * t] = v; o[2 * t + 1] = v;
}

__global__ void cm_to_tok(const float* __restrict__ in, float* __restrict__ out,
                          int C, int T, int nB)  // (B,C,T) -> (B*T, C)
{
  int idx = blockIdx.x * blockDim.x + threadIdx.x;
  if (idx >= nB * C * T) return;
  int t = idx % T; int c = (idx / T) % C; int b = idx / (T * C);
  out[((long)b * T + t) * C + c] = in[((long)b * C + c) * T + t];
}

__global__ void tok_to_cm(const float* __restrict__ in, float* __restrict__ out,
                          int C, int T, int nB)  // (B*T, C) -> (B,C,T)
{
  int idx = blockIdx.x * blockDim.x + threadIdx.x;
  if (idx >= nB * C * T) return;
  int t = idx % T; int c = (idx / T) % C; int b = idx / (T * C);
  out[((long)b * C + c) * T + t] = in[((long)b * T + t) * C + c];
}

__global__ void row_sumsq(const float* __restrict__ E, float* __restrict__ out,
                          int N, int Kd)
{
  int row  = (blockIdx.x * blockDim.x + threadIdx.x) >> 5;
  int lane = threadIdx.x & 31;
  if (row >= N) return;
  const float* p = E + (long)row * Kd;
  float s = 0.f;
  for (int j = lane; j < Kd; j += 32) { float v = p[j]; s += v * v; }
  for (int o = 16; o; o >>= 1) s += __shfl_xor(s, o, 32);
  if (!lane) out[row] = s;
}

// ---- VQ: argmin(||z-e||) via (|e|^2 - 2 z.e), gather zq, loss partial ------
__global__ void vq_select(const float* __restrict__ G, const float* __restrict__ en,
                          const float* __restrict__ H, const float* __restrict__ cb,
                          float* __restrict__ ZQ, float* __restrict__ idx_out,
                          float* __restrict__ lossAcc, int Mrows)
{
  int row  = (blockIdx.x * blockDim.x + threadIdx.x) >> 5;
  int lane = threadIdx.x & 31;
  if (row >= Mrows) return;
  const float* g = G + (long)row * 1024;
  float best = 3.4e38f; int bi = 0;
  for (int j = lane; j < 1024; j += 32) {
    float d = en[j] - 2.f * g[j];
    if (d < best) { best = d; bi = j; }
  }
  for (int o = 16; o; o >>= 1) {
    float ob = __shfl_xor(best, o, 32);
    int   oi = __shfl_xor(bi, o, 32);
    if (ob < best || (ob == best && oi < bi)) { best = ob; bi = oi; }
  }
  if (!lane) idx_out[row] = (float)bi;
  const float* e = cb + (long)bi * 768;
  const float* h = H + (long)row * 768;
  float* z = ZQ + (long)row * 768;
  float ls = 0.f;
#pragma unroll
  for (int i = 0; i < 24; ++i) {
    int c = lane + 32 * i;
    float ev = e[c]; z[c] = ev;
    float d = ev - h[c]; ls += d * d;
  }
  for (int o = 16; o; o >>= 1) ls += __shfl_xor(ls, o, 32);
  if (!lane) atomicAdd(lossAcc, ls);
}

__global__ void zero1(float* p) { if (blockIdx.x == 0 && threadIdx.x == 0) *p = 0.f; }
__global__ void loss_fin(const float* acc, float* out)
{ if (blockIdx.x == 0 && threadIdx.x == 0) *out = *acc * (1.25f / (256.f * 768.f)); }

// ---------------------------------------------------------------------------
extern "C" void kernel_launch(void* const* d_in, const int* in_sizes, int n_in,
                              void* d_out, int out_size, void* d_ws, size_t ws_size,
                              hipStream_t stream)
{
  (void)in_sizes; (void)n_in; (void)out_size; (void)ws_size;
  auto P = [&](int i) { return (const float*)d_in[i]; };
  // setup_inputs() dict order
  const float* x        = P(0);   const float* codebook = P(1);
  const float* ec0w = P(2), *ec0b = P(3), *ebn0g = P(4), *ebn0b = P(5);
  const float* ecw  = P(6), *ecb  = P(7), *ebng  = P(8), *ebnb  = P(9);
  const float* elinw = P(10), *elinb = P(11);
  const float* einw = P(12), *einb = P(13), *eoutw = P(14), *eoutb = P(15);
  const float* eln1g = P(16), *eln1b = P(17);
  const float* ef1w = P(18), *ef1b = P(19), *ef2w = P(20), *ef2b = P(21);
  const float* eln2g = P(22), *eln2b = P(23);
  const float* dinw = P(24), *dinb = P(25), *doutw = P(26), *doutb = P(27);
  const float* dln1g = P(28), *dln1b = P(29);
  const float* df1w = P(30), *df1b = P(31), *df2w = P(32), *df2b = P(33);
  const float* dln2g = P(34), *dln2b = P(35);
  const float* dctw = P(36), *dctb = P(37), *dbn0g = P(38), *dbn0b = P(39);
  const float* dcw = P(40), *dcb = P(41), *dbng = P(42), *dbnb = P(43);
  const float* dlinw = P(44), *dlinb = P(45);
  const float* crw = P(46), *crb = P(47);

  float* out = (float*)d_out;
  const int OUT_DEC = 8 * 256 * 53;      // 108544 floats, then loss, then idx

  // workspace arena (floats)
  float* ws = (float*)d_ws;
  long o = 0;
  float* bufA  = ws + o; o += 8L * 768 * 256;
  float* bufB  = ws + o; o += 8L * 768 * 256;
  float* tok0  = ws + o; o += 2048L * 768;
  float* tok1  = ws + o; o += 2048L * 768;
  float* tok2  = ws + o; o += 2048L * 768;
  float* qkvb  = ws + o; o += 2048L * 2304;
  float* sbuf  = ws + o; o += 96L * 256 * 256;
  float* ffb   = ws + o; o += 2048L * 3072;
  float* gbuf  = ws + o; o += 256L * 1024;
  float* enorm = ws + o; o += 1024;
  float* lacc  = ws + o; o += 16;

  auto cdiv = [](long a, long b) { return (int)((a + b - 1) / b); };
  auto gemm = [&](const float* A, const float* Bt, const float* bias, float* C,
                  int M, int N, int K, int act) {
    int waves = (M >> 6) * (N >> 6);             // 64x64 per wave
    gemm_bias_act<<<cdiv((long)waves * 32, 256), 256, 0, stream>>>(A, Bt, bias, C, M, N, K, act);
  };
  auto transformer = [&](float* X, float* Ob, float* Zb, int M, int Tb,
                         const float* inw, const float* inb, const float* ow, const float* ob,
                         const float* g1, const float* b1, const float* f1w, const float* f1b,
                         const float* f2w, const float* f2b, const float* g2, const float* b2) {
    for (int l = 0; l < 4; ++l) {
      gemm(X, inw + (long)l * 2304 * 768, inb + l * 2304, qkvb, M, 2304, 768, 0);
      { int tl = (Tb / 16) * (Tb / 16);
        dim3 gr(cdiv((long)tl * 32, 256), 1, 96);
        attn_scores<<<gr, 256, 0, stream>>>(qkvb, sbuf, Tb); }
      { int rows = 96 * Tb;
        softmax_rows<<<cdiv((long)rows * 32, 256), 256, 0, stream>>>(sbuf, Tb, rows); }
      { int tl = (Tb / 16) * 4;
        dim3 gr(cdiv((long)tl * 32, 256), 1, 96);
        attn_av<<<gr, 256, 0, stream>>>(sbuf, qkvb, Ob, Tb); }
      gemm(Ob, ow + (long)l * 768 * 768, ob + l * 768, Zb, M, 768, 768, 0);
      ln_add_768<<<cdiv((long)M * 32, 256), 256, 0, stream>>>(X, Zb, g1 + l * 768, b1 + l * 768, X, M);
      gemm(X, f1w + (long)l * 3072 * 768, f1b + l * 3072, ffb, M, 3072, 768, 1);
      gemm(ffb, f2w + (long)l * 768 * 3072, f2b + l * 768, Zb, M, 768, 3072, 0);
      ln_add_768<<<cdiv((long)M * 32, 256), 256, 0, stream>>>(X, Zb, g2 + l * 768, b2 + l * 768, X, M);
    }
  };

  // ======================= encoder squasher =======================
  // conv0: x (B,256,53) read as (B,53,256) via strides; stride2 -> (B,768,128)
  conv1d_k5<<<cdiv(8L * 768 * 128, 256), 256, 0, stream>>>(
      x, 256L * 53, 1, 53, 53, 256, ec0w, ec0b, ebn0g, ebn0b,
      bufA, 768L * 128, 128, 1, 768, 128, 2, 0, 8);
  // i=0: conv @128 -> pool 64
  conv1d_k5<<<cdiv(8L * 768 * 128, 256), 256, 0, stream>>>(
      bufA, 768L * 128, 128, 1, 768, 128, ecw, ecb, ebng, ebnb,
      bufB, 768L * 128, 128, 1, 768, 128, 1, 0, 8);
  maxpool2<<<cdiv(8L * 768 * 64, 256), 256, 0, stream>>>(bufB, bufA, 768, 64, 8);
  // i=1: conv @64 -> pool 32
  conv1d_k5<<<cdiv(8L * 768 * 64, 256), 256, 0, stream>>>(
      bufA, 768L * 64, 64, 1, 768, 64, ecw + 768L * 768 * 5, ecb + 768, ebng + 768, ebnb + 768,
      bufB, 768L * 64, 64, 1, 768, 64, 1, 0, 8);
  maxpool2<<<cdiv(8L * 768 * 32, 256), 256, 0, stream>>>(bufB, bufA, 768, 32, 8);
  cm_to_tok<<<cdiv(8L * 768 * 32, 256), 256, 0, stream>>>(bufA, tok0, 768, 32, 8);

  // enc linear + transformer (256 tokens, Tb=32)
  gemm(tok0, elinw, elinb, tok1, 256, 768, 768, 0);
  transformer(tok1, tok0, tok2, 256, 32, einw, einb, eoutw, eoutb,
              eln1g, eln1b, ef1w, ef1b, ef2w, ef2b, eln2g, eln2b);

  // ======================= vector quantization =======================
  row_sumsq<<<cdiv(1024L * 32, 256), 256, 0, stream>>>(codebook, enorm, 1024, 768);
  gemm(tok1, codebook, nullptr, gbuf, 256, 1024, 768, 0);     // z @ e^T (WMMA)
  zero1<<<1, 32, 0, stream>>>(lacc);
  vq_select<<<cdiv(256L * 32, 256), 256, 0, stream>>>(
      gbuf, enorm, tok1, codebook, tok0, out + OUT_DEC + 1, lacc, 256);
  loss_fin<<<1, 32, 0, stream>>>(lacc, out + OUT_DEC);

  // ======================= decoder expander =======================
  tok_to_cm<<<cdiv(8L * 768 * 32, 256), 256, 0, stream>>>(tok0, bufA, 768, 32, 8);
  convT_k5_s2<<<cdiv(8L * 768 * 64, 256), 256, 0, stream>>>(
      bufA, 768, 32, dctw, dctb, dbn0g, dbn0b, bufB, 768, 64, 8);
  // i=0: conv @64 -> repeat 128
  conv1d_k5<<<cdiv(8L * 768 * 64, 256), 256, 0, stream>>>(
      bufB, 768L * 64, 64, 1, 768, 64, dcw, dcb, dbng, dbnb,
      bufA, 768L * 64, 64, 1, 768, 64, 1, 0, 8);
  repeat2<<<cdiv(8L * 768 * 64, 256), 256, 0, stream>>>(bufA, bufB, 768, 64, 8);
  // i=1: conv @128 -> repeat 256
  conv1d_k5<<<cdiv(8L * 768 * 128, 256), 256, 0, stream>>>(
      bufB, 768L * 128, 128, 1, 768, 128, dcw + 768L * 768 * 5, dcb + 768, dbng + 768, dbnb + 768,
      bufA, 768L * 128, 128, 1, 768, 128, 1, 0, 8);
  repeat2<<<cdiv(8L * 768 * 128, 256), 256, 0, stream>>>(bufA, bufB, 768, 128, 8);
  cm_to_tok<<<cdiv(8L * 768 * 256, 256), 256, 0, stream>>>(bufB, tok0, 768, 256, 8);

  // dec linear + transformer (2048 tokens, Tb=256)
  gemm(tok0, dlinw, dlinb, tok1, 2048, 768, 768, 0);
  transformer(tok1, tok0, tok2, 2048, 256, dinw, dinb, doutw, doutb,
              dln1g, dln1b, df1w, df1b, df2w, df2b, dln2g, dln2b);

  // cross conv (zero pad, no act/BN), write transposed straight into d_out
  tok_to_cm<<<cdiv(8L * 768 * 256, 256), 256, 0, stream>>>(tok1, bufA, 768, 256, 8);
  conv1d_k5<<<cdiv(8L * 53 * 256, 256), 256, 0, stream>>>(
      bufA, 768L * 256, 256, 1, 768, 256, crw, crb, nullptr, nullptr,
      out, 256L * 53, 1, 53, 53, 256, 1, 1, 8);
}
